// MSEAAggregation_32521492365734
// MI455X (gfx1250) — matrix operand order repaired and checked by reading, not compiled, not run on hardware
//
#include <hip/hip_runtime.h>

typedef __attribute__((ext_vector_type(16))) __bf16 v16bf;
typedef __attribute__((ext_vector_type(8)))  __bf16 v8bf;
typedef __attribute__((ext_vector_type(8)))  float  v8f;

#define V_NODES 50000
#define E_EDGES 800000
#define D_FEAT  128      // node feature dim; EH = NH = 128
#define WPB     4        // waves per block (wave32)

// ---------------------------------------------------------------------------
// Prep kernels
// ---------------------------------------------------------------------------
__global__ void f32_to_bf16_kernel(const float* __restrict__ in,
                                   __bf16* __restrict__ out, int n) {
    int i = blockIdx.x * blockDim.x + threadIdx.x;
    if (i < n) out[i] = (__bf16)in[i];
}

__global__ void zero_f32_kernel(float* __restrict__ p, int n) {
    int i = blockIdx.x * blockDim.x + threadIdx.x;
    if (i < n) p[i] = 0.0f;
}

// Pack W (K x N, row-major fp32) into WMMA B-fragment order (bf16):
// frag f = (nt*(K/32)+kb), lane l, element e:
//   lanes 0-15  : column nt*16+l,     K = kb*32 + (e<8 ? e : 16+e-8)
//   lanes 16-31 : column nt*16+(l-16),K = kb*32 + 8 + (e<8 ? e : 16+e-8)
__global__ void pack_weight_kernel(const float* __restrict__ W,
                                   __bf16* __restrict__ Wp, int K, int N) {
    int idx = blockIdx.x * blockDim.x + threadIdx.x;
    if (idx >= K * N) return;
    int e    = idx & 15;
    int lane = (idx >> 4) & 31;
    int kb   = (idx >> 9) % (K >> 5);
    int nt   = (idx >> 9) / (K >> 5);
    int k = kb * 32 + ((e >= 8) ? 16 : 0) + ((lane >= 16) ? 8 : 0) + (e & 7);
    int n = nt * 16 + (lane & 15);
    Wp[idx] = (__bf16)W[k * N + n];
}

// ---------------------------------------------------------------------------
// Shared helpers (device)
// ---------------------------------------------------------------------------
__device__ __forceinline__ v16bf load_a_frag(const __bf16* row_ptr) {
    // elements 0..7 = K+0..7 , elements 8..15 = K+16..23 (per-lane half offset
    // already applied by caller) -> two ds_load_b128
    v8bf lo = *(const v8bf*)(row_ptr);
    v8bf hi = *(const v8bf*)(row_ptr + 16);
    return __builtin_shufflevector(lo, hi, 0,1,2,3,4,5,6,7,8,9,10,11,12,13,14,15);
}

__device__ __forceinline__ v8f splat8(float x) {
    v8f r = {x, x, x, x, x, x, x, x};
    return r;
}

// ---------------------------------------------------------------------------
// Edge MLP: per wave, a 16-edge tile.
//   L1: (16x256)@(256x128)  L2: (16x128)@(128x128)  L3: (16x128)@(128x128)
// then scatter-add messages into agg via hardware f32 atomics.
// ---------------------------------------------------------------------------
__global__ __launch_bounds__(WPB * 32)
void edge_mlp_kernel(const __bf16* __restrict__ hbf,
                     const int*    __restrict__ ei,
                     const __bf16* __restrict__ W1p, const float* __restrict__ b1,
                     const __bf16* __restrict__ W2p, const float* __restrict__ b2,
                     const __bf16* __restrict__ W3p, const float* __restrict__ b3,
                     float* __restrict__ agg)
{
    __shared__ __bf16 sA[WPB][16 * 256];   // gathered edge features / L2 out
    __shared__ __bf16 sM[WPB][16 * 128];   // L1 output

    const int lane = threadIdx.x & 31;
    const int wave = threadIdx.x >> 5;
    const int base = (blockIdx.x * WPB + wave) * 16;   // exact: E = 12500*64

    __bf16* A = sA[wave];
    __bf16* M = sM[wave];
    const int* srcArr = ei;
    const int* dstArr = ei + E_EDGES;

    // ---- gather: 16 rows x 256 bf16, 8-element chunks, coalesced per wave
    for (int c = lane; c < 512; c += 32) {
        int row  = c >> 5;
        int off  = (c & 31) * 8;                     // 0..248
        int edge = base + row;
        int node = (off < 128) ? srcArr[edge] : dstArr[edge];
        v8bf chunk = *(const v8bf*)(hbf + (size_t)node * 128 + (off & 127));
        *(v8bf*)(A + row * 256 + off) = chunk;
    }
    __syncthreads();

    const int hi16 = (lane >= 16) ? 1 : 0;
    const int rlo  = lane & 15;

    // ---- layer 1: K=256 (8 k-blocks), N=128 (8 n-tiles)
    #pragma unroll
    for (int nt = 0; nt < 8; ++nt) {
        v8f acc = splat8(b1[nt * 16 + rlo]);
        #pragma unroll
        for (int kb = 0; kb < 8; ++kb) {
            v16bf a = load_a_frag(A + rlo * 256 + kb * 32 + hi16 * 8);
            v16bf b = *(const v16bf*)(W1p + (((nt << 3) + kb) * 32 + lane) * 16);
            acc = __builtin_amdgcn_wmma_f32_16x16x32_bf16(false, a, false, b,
                                                          (short)0, acc, false, false);
        }
        int col = nt * 16 + rlo;
        #pragma unroll
        for (int r = 0; r < 8; ++r) {
            float v = acc[r] > 0.0f ? acc[r] : 0.0f;
            M[(r + hi16 * 8) * 128 + col] = (__bf16)v;
        }
    }
    __syncthreads();

    // ---- layer 2: K=128 (4 k-blocks); output (relu) reuses A region
    #pragma unroll
    for (int nt = 0; nt < 8; ++nt) {
        v8f acc = splat8(b2[nt * 16 + rlo]);
        #pragma unroll
        for (int kb = 0; kb < 4; ++kb) {
            v16bf a = load_a_frag(M + rlo * 128 + kb * 32 + hi16 * 8);
            v16bf b = *(const v16bf*)(W2p + (((nt << 2) + kb) * 32 + lane) * 16);
            acc = __builtin_amdgcn_wmma_f32_16x16x32_bf16(false, a, false, b,
                                                          (short)0, acc, false, false);
        }
        int col = nt * 16 + rlo;
        #pragma unroll
        for (int r = 0; r < 8; ++r) {
            float v = acc[r] > 0.0f ? acc[r] : 0.0f;
            A[(r + hi16 * 8) * 128 + col] = (__bf16)v;
        }
    }
    __syncthreads();

    // ---- layer 3: K=128, then scatter-add into agg[dst]
    #pragma unroll
    for (int nt = 0; nt < 8; ++nt) {
        v8f acc = splat8(b3[nt * 16 + rlo]);
        #pragma unroll
        for (int kb = 0; kb < 4; ++kb) {
            v16bf a = load_a_frag(A + rlo * 128 + kb * 32 + hi16 * 8);
            v16bf b = *(const v16bf*)(W3p + (((nt << 2) + kb) * 32 + lane) * 16);
            acc = __builtin_amdgcn_wmma_f32_16x16x32_bf16(false, a, false, b,
                                                          (short)0, acc, false, false);
        }
        int col = nt * 16 + rlo;
        #pragma unroll
        for (int r = 0; r < 8; ++r) {
            int row = r + hi16 * 8;
            int d   = dstArr[base + row];
            unsafeAtomicAdd(agg + (size_t)d * 128 + col, acc[r]);
        }
    }
}

// ---------------------------------------------------------------------------
// Node MLP: per wave, a 16-node tile on x = [h | agg]  (16x256)
//   L1: (16x256)@(256x128) relu   L2: (16x128)@(128x128) -> out
// ---------------------------------------------------------------------------
__global__ __launch_bounds__(WPB * 32)
void node_mlp_kernel(const __bf16* __restrict__ hbf,
                     const float*  __restrict__ agg,
                     const __bf16* __restrict__ Wn1p, const float* __restrict__ bn1,
                     const __bf16* __restrict__ Wn2p, const float* __restrict__ bn2,
                     float* __restrict__ out)
{
    __shared__ __bf16 sA[WPB][16 * 256];
    __shared__ __bf16 sM[WPB][16 * 128];

    const int lane = threadIdx.x & 31;
    const int wave = threadIdx.x >> 5;
    const int base = (blockIdx.x * WPB + wave) * 16;

    __bf16* A = sA[wave];
    __bf16* M = sM[wave];

    // ---- build input tile: cols 0-127 = hbf[v], cols 128-255 = bf16(agg[v])
    for (int c = lane; c < 512; c += 32) {
        int row = c >> 5;
        int off = (c & 31) * 8;
        int v   = base + row;
        if (v >= V_NODES) v = V_NODES - 1;          // clamp (stores are guarded)
        if (off < 128) {
            *(v8bf*)(A + row * 256 + off) =
                *(const v8bf*)(hbf + (size_t)v * 128 + off);
        } else {
            v8f af = *(const v8f*)(agg + (size_t)v * 128 + (off - 128));
            v8bf t;
            #pragma unroll
            for (int i = 0; i < 8; ++i) t[i] = (__bf16)af[i];
            *(v8bf*)(A + row * 256 + off) = t;
        }
    }
    __syncthreads();

    const int hi16 = (lane >= 16) ? 1 : 0;
    const int rlo  = lane & 15;

    // ---- layer n1: K=256, relu -> M
    #pragma unroll
    for (int nt = 0; nt < 8; ++nt) {
        v8f acc = splat8(bn1[nt * 16 + rlo]);
        #pragma unroll
        for (int kb = 0; kb < 8; ++kb) {
            v16bf a = load_a_frag(A + rlo * 256 + kb * 32 + hi16 * 8);
            v16bf b = *(const v16bf*)(Wn1p + (((nt << 3) + kb) * 32 + lane) * 16);
            acc = __builtin_amdgcn_wmma_f32_16x16x32_bf16(false, a, false, b,
                                                          (short)0, acc, false, false);
        }
        int col = nt * 16 + rlo;
        #pragma unroll
        for (int r = 0; r < 8; ++r) {
            float v = acc[r] > 0.0f ? acc[r] : 0.0f;
            M[(r + hi16 * 8) * 128 + col] = (__bf16)v;
        }
    }
    __syncthreads();

    // ---- layer n2: K=128 -> fp32 output
    #pragma unroll
    for (int nt = 0; nt < 8; ++nt) {
        v8f acc = splat8(bn2[nt * 16 + rlo]);
        #pragma unroll
        for (int kb = 0; kb < 4; ++kb) {
            v16bf a = load_a_frag(M + rlo * 128 + kb * 32 + hi16 * 8);
            v16bf b = *(const v16bf*)(Wn2p + (((nt << 2) + kb) * 32 + lane) * 16);
            acc = __builtin_amdgcn_wmma_f32_16x16x32_bf16(false, a, false, b,
                                                          (short)0, acc, false, false);
        }
        int col = nt * 16 + rlo;
        #pragma unroll
        for (int r = 0; r < 8; ++r) {
            int v = base + r + hi16 * 8;
            if (v < V_NODES) out[(size_t)v * 128 + col] = acc[r];
        }
    }
}

// ---------------------------------------------------------------------------
// Launch
// ---------------------------------------------------------------------------
extern "C" void kernel_launch(void* const* d_in, const int* in_sizes, int n_in,
                              void* d_out, int out_size, void* d_ws, size_t ws_size,
                              hipStream_t stream) {
    const float* h   = (const float*)d_in[0];
    const int*   ei  = (const int*)  d_in[1];
    const float* W1  = (const float*)d_in[2];
    const float* b1  = (const float*)d_in[3];
    const float* W2  = (const float*)d_in[4];
    const float* b2  = (const float*)d_in[5];
    const float* W3  = (const float*)d_in[6];
    const float* b3  = (const float*)d_in[7];
    const float* Wn1 = (const float*)d_in[8];
    const float* bn1 = (const float*)d_in[9];
    const float* Wn2 = (const float*)d_in[10];
    const float* bn2 = (const float*)d_in[11];
    float* out = (float*)d_out;

    char* ws = (char*)d_ws;
    size_t off = 0;
    __bf16* hbf  = (__bf16*)(ws + off); off += (size_t)V_NODES * 128 * 2;
    float*  agg  = (float*) (ws + off); off += (size_t)V_NODES * 128 * 4;
    __bf16* W1p  = (__bf16*)(ws + off); off += (size_t)256 * 128 * 2;
    __bf16* W2p  = (__bf16*)(ws + off); off += (size_t)128 * 128 * 2;
    __bf16* W3p  = (__bf16*)(ws + off); off += (size_t)128 * 128 * 2;
    __bf16* Wn1p = (__bf16*)(ws + off); off += (size_t)256 * 128 * 2;
    __bf16* Wn2p = (__bf16*)(ws + off); off += (size_t)128 * 128 * 2;

    const int hv = V_NODES * 128;
    f32_to_bf16_kernel<<<(hv + 255) / 256, 256, 0, stream>>>(h, hbf, hv);
    zero_f32_kernel   <<<(hv + 255) / 256, 256, 0, stream>>>(agg, hv);
    pack_weight_kernel<<<(256 * 128 + 255) / 256, 256, 0, stream>>>(W1,  W1p,  256, 128);
    pack_weight_kernel<<<(128 * 128 + 255) / 256, 256, 0, stream>>>(W2,  W2p,  128, 128);
    pack_weight_kernel<<<(128 * 128 + 255) / 256, 256, 0, stream>>>(W3,  W3p,  128, 128);
    pack_weight_kernel<<<(256 * 128 + 255) / 256, 256, 0, stream>>>(Wn1, Wn1p, 256, 128);
    pack_weight_kernel<<<(128 * 128 + 255) / 256, 256, 0, stream>>>(Wn2, Wn2p, 128, 128);

    // E = 800000 = 12500 blocks * 4 waves * 16 edges (exact)
    edge_mlp_kernel<<<E_EDGES / (WPB * 16), WPB * 32, 0, stream>>>(
        hbf, ei, W1p, b1, W2p, b2, W3p, b3, agg);

    // V = 50000 -> 3125 tiles -> 782 blocks (tail guarded)
    int node_tiles  = (V_NODES + 15) / 16;
    int node_blocks = (node_tiles + WPB - 1) / WPB;
    node_mlp_kernel<<<node_blocks, WPB * 32, 0, stream>>>(
        hbf, agg, Wn1p, bn1, Wn2p, bn2, out);
}
